// CNN2D_SE_Kagome_108_20633022890677
// MI455X (gfx1250) — matrix operand order repaired and checked by reading, not compiled, not run on hardware
//
#include <hip/hip_runtime.h>

// ---------------------------------------------------------------------------
// Types for gfx1250 WMMA
// ---------------------------------------------------------------------------
typedef __attribute__((ext_vector_type(16))) __bf16 v16bf;
typedef __attribute__((ext_vector_type(8)))  __bf16 v8bf;
typedef __attribute__((ext_vector_type(4)))  __bf16 v4bf;
typedef __attribute__((ext_vector_type(8)))  float  v8f;

// ---------------------------------------------------------------------------
// Compile-time tables: periodic-boundary pixel map, per-row-tile slot tables,
// output mask bits.
// ---------------------------------------------------------------------------
struct KTab {
    short         srcpix[8][73];     // row-tile, slot -> source pixel (-1 = zero)
    unsigned char cslot[8][4][18];   // row-tile, kh, padded-col -> slot
    unsigned int  maskbits[16];      // OUT_MASK, bit x of row y
};

constexpr KTab make_tab() {
    KTab T{};
    // padded 18x18 -> source pixel in original 16x16 (or -1 = zero)
    int pix[18][18]{};
    for (int r = 0; r < 18; ++r)
        for (int c = 0; c < 18; ++c)
            pix[r][c] = (r >= 1 && r <= 16 && c >= 1 && c <= 16) ? (r-1)*16 + (c-1) : -1;
    const int DR[30] = {1,1,2,3,4,4,6,7,8,10,11,12,14,14,15,16,17,17,16,15,14,14,12,10,8,6,4,4,3,2};
    const int DC[30] = {3,5,7,9,10,11,13,13,14,15,15,16,15,16,15,14,13,11,9,7,6,5,3,2,1,0,0,1,1,2};
    const int SR[30] = {13,13,14,15,16,16,6,7,8,10,11,12,2,2,3,4,5,5,4,3,2,2,12,10,8,6,16,16,15,14};
    const int SC[30] = {15,5,7,9,10,11,1,1,2,3,3,4,3,4,3,2,1,11,9,7,6,5,15,14,13,12,12,13,13,14};
    for (int k = 0; k < 30; ++k) pix[DR[k]][DC[k]] = (SR[k]-1)*16 + (SC[k]-1);

    // per output-row tile i: compact slots for padded rows 2i..2i+3, cols 0..17
    for (int i = 0; i < 8; ++i) {
        for (int s = 0; s < 73; ++s) T.srcpix[i][s] = -1;
        int cnt = 0;
        for (int kh = 0; kh < 4; ++kh)
            for (int cc = 0; cc < 18; ++cc) {
                const int p = pix[2*i + kh][cc];
                int slot;
                if (p < 0) slot = 72;                      // zero slot
                else {
                    slot = -1;
                    for (int s = 0; s < cnt; ++s)
                        if (T.srcpix[i][s] == p) { slot = s; break; }
                    if (slot < 0) { slot = cnt; T.srcpix[i][cnt] = (short)p; ++cnt; }
                }
                T.cslot[i][kh][cc] = (unsigned char)slot;
            }
    }
    // OUT_MASK
    int m[16][16]{};
    for (int r = 0; r < 16; ++r) for (int c = 0; c < 16; ++c) m[r][c] = 1;
    for (int i = 0; i < 9; ++i) for (int c = 7+i; c < 16; ++c) m[i][c] = 0;
    for (int i = 0; i < 7; ++i) for (int c = 0; c <= i; ++c) m[9+i][c] = 0;
    m[0][4]=0; m[0][5]=0; m[0][6]=0;
    m[1][6]=0; m[1][7]=0;
    m[2][8]=0; m[3][9]=0;
    m[6][12]=0; m[7][13]=0; m[8][14]=0; m[9][14]=0; m[10][14]=0; m[11][15]=0;
    for (int r = 13; r < 16; ++r) { m[r][14]=0; m[r][15]=0; }
    m[15][13]=0; m[15][7]=0; m[15][8]=0;
    m[13][5]=0; m[14][6]=0;
    m[8][0]=0; m[9][1]=0; m[7][0]=0; m[3][0]=0;
    for (int r = 0; r < 3; ++r) { m[r][0]=0; m[r][1]=0; }
    m[0][2]=0;
    for (int r = 0; r < 16; ++r) {
        unsigned bits = 0;
        for (int c = 0; c < 16; ++c) if (m[r][c]) bits |= (1u << c);
        T.maskbits[r] = bits;
    }
    return T;
}

__constant__ KTab TAB = make_tab();

// tap lists (5 real taps + 1 zero-pad tap); (kh,kw) per conv {up,left,right}
__constant__ int cKH[3][6] = {{0,0,1,2,2,0},{1,2,2,2,3,0},{1,2,2,2,3,0}};
__constant__ int cKW[3][6] = {{0,1,1,1,2,0},{1,0,1,2,1,0},{1,1,2,3,3,0}};

// ---------------------------------------------------------------------------
// Weight prep: fp32 dense [O=64][C=64][4][4] -> bf16 WMMA-B layout in d_ws:
//   Wg[(((d*3+tp)*4+cb)*64 + o)*32 + k],  k: 0..15 -> tap 2tp, c = cb*16+k
//                                          k: 16..31 -> tap 2tp+1, c = cb*16+k-16
// Only the 5 sparse taps are non-zero (mask applied implicitly).
// ---------------------------------------------------------------------------
__global__ __launch_bounds__(256) void kagome_prep_weights(
    const float* __restrict__ wu, const float* __restrict__ wl,
    const float* __restrict__ wr, __bf16* __restrict__ Wg)
{
    const int idx = blockIdx.x * 256 + threadIdx.x;
    if (idx >= 3*3*4*64*32) return;
    const int k  = idx & 31;
    const int o  = (idx >> 5) & 63;
    const int cb = (idx >> 11) & 3;
    const int dt = idx >> 13;          // d*3+tp, 0..8
    const int tp = dt % 3;
    const int d  = dt / 3;
    const int t  = 2*tp + (k >> 4);
    const int c  = cb*16 + (k & 15);
    float v = 0.0f;
    if (t < 5) {
        const float* w = (d == 0) ? wu : ((d == 1) ? wl : wr);
        v = w[((o*64 + c)*4 + cKH[d][t])*4 + cKW[d][t]];
    }
    Wg[idx] = (__bf16)v;
}

// ---------------------------------------------------------------------------
// Main kernel: one workgroup = (8 batches) x (1 output row i).
// 8 waves: wave w -> o-block (w&3), position-half (w>>2).
// WMMA tile: M = {posA b0..7, posB b0..7}, N = 16 out-channels,
//            K = 32 = (tap-pair) x (16 channels).
// ---------------------------------------------------------------------------
__global__ __launch_bounds__(256) void kagome_main(
    const float* __restrict__ x,  const float* __restrict__ bu,
    const float* __restrict__ bl, const float* __restrict__ br,
    const __bf16* __restrict__ Wg, float* __restrict__ out)
{
    // Xs: bf16 [73 slots][8 b][72 c]  (84096 B)   -- input staging
    // Os: f32  [8 b][64 o][36]        (73728 B)   -- output staging (reuses Xs)
    __shared__ __align__(16) unsigned char smem_raw[84096];
    __bf16* Xs = reinterpret_cast<__bf16*>(smem_raw);
    float*  Os = reinterpret_cast<float*>(smem_raw);

    const int tid  = threadIdx.x;
    const int b0   = blockIdx.x * 8;
    const int irow = blockIdx.y;

    // ---- stage x (fp32 global, coalesced over pixels) -> bf16 LDS ----
    for (int idx = tid; idx < 73*128; idx += 256) {
        const int s   = idx % 73;          // slot (lane-fastest => coalesced px)
        const int rem = idx / 73;
        const int b   = rem & 7;
        const int qd  = rem >> 3;          // channel quad 0..15
        const int p   = TAB.srcpix[irow][s];
        float f0 = 0.f, f1 = 0.f, f2 = 0.f, f3 = 0.f;
        if (p >= 0) {
            const float* src = x + (size_t)(b0 + b)*16384 + (size_t)(qd*4)*256 + p;
            f0 = src[0]; f1 = src[256]; f2 = src[512]; f3 = src[768];
        }
        v4bf h;
        h[0] = (__bf16)f0; h[1] = (__bf16)f1; h[2] = (__bf16)f2; h[3] = (__bf16)f3;
        *reinterpret_cast<v4bf*>(&Xs[(s*8 + b)*72 + qd*4]) = h;   // ds_store_b64
    }
    __syncthreads();

    const int lane   = tid & 31;
    const int wv     = tid >> 5;
    const int nb     = wv & 3;            // o-block
    const int hf     = wv >> 2;           // position half
    const int oc     = nb*16 + (lane & 15);
    const int bsel   = lane & 7;          // A-row batch
    const int posSel = (lane >> 3) & 1;   // A-row position within pair
    const int khalf  = lane >> 4;         // K half (c offset 0/8)

    const float biasv[3] = { bu[oc], bl[oc], br[oc] };

    v8f acc[6];
    #pragma unroll
    for (int d = 0; d < 3; ++d)
        #pragma unroll
        for (int q = 0; q < 2; ++q)
            #pragma unroll
            for (int r = 0; r < 8; ++r)
                acc[d*2 + q][r] = biasv[d];

    #pragma unroll
    for (int d = 0; d < 3; ++d) {
        #pragma unroll
        for (int tp = 0; tp < 3; ++tp) {
            int aoff[2][2];
            #pragma unroll
            for (int q = 0; q < 2; ++q) {
                const int j = 2*(hf*2 + q) + posSel;      // output column
                #pragma unroll
                for (int tt = 0; tt < 2; ++tt) {
                    const int t = 2*tp + tt;
                    const int slot = (t < 5)
                        ? (int)TAB.cslot[irow][cKH[d][t]][2*j + cKW[d][t]]
                        : 72;
                    aoff[q][tt] = (slot*8 + bsel)*72 + khalf*8;
                }
            }
            const __bf16* bp = Wg + ((size_t)((d*3 + tp)*4)*64 + oc)*32 + khalf*8;
            #pragma unroll
            for (int cb = 0; cb < 4; ++cb) {
                const v8bf bq0 = *reinterpret_cast<const v8bf*>(bp + (size_t)cb*2048);
                const v8bf bq1 = *reinterpret_cast<const v8bf*>(bp + (size_t)cb*2048 + 16);
                const v16bf bf = __builtin_shufflevector(bq0, bq1,
                    0,1,2,3,4,5,6,7,8,9,10,11,12,13,14,15);
                #pragma unroll
                for (int q = 0; q < 2; ++q) {
                    const v8bf a0 = *reinterpret_cast<const v8bf*>(Xs + aoff[q][0] + cb*16);
                    const v8bf a1 = *reinterpret_cast<const v8bf*>(Xs + aoff[q][1] + cb*16);
                    const v16bf af = __builtin_shufflevector(a0, a1,
                        0,1,2,3,4,5,6,7,8,9,10,11,12,13,14,15);
                    acc[d*2 + q] = __builtin_amdgcn_wmma_f32_16x16x32_bf16(
                        false, af, false, bf, (short)0, acc[d*2 + q], false, false);
                }
            }
        }
    }

    __syncthreads();   // done with Xs; reuse LDS for Os

    // ---- masked interleave into LDS output tile ----
    #pragma unroll
    for (int d = 0; d < 3; ++d) {
        const int yloc = (d > 0) ? 1 : 0;
        const unsigned mrow = TAB.maskbits[2*irow + yloc];
        #pragma unroll
        for (int q = 0; q < 2; ++q) {
            const int j    = 2*(hf*2 + q) + khalf;        // C-tile: posSel = lane>>4
            const int xcol = 2*j + ((d == 2) ? 1 : 0);
            const float mv = ((mrow >> xcol) & 1u) ? 1.0f : 0.0f;
            #pragma unroll
            for (int r = 0; r < 8; ++r) {
                const int base = (r*64 + oc)*36 + yloc*16;
                Os[base + xcol] = acc[d*2 + q][r] * mv;
                if (d == 0) Os[base + xcol + 1] = 0.0f;   // even row, odd col = 0
            }
        }
    }
    __syncthreads();

    // ---- coalesced 128B-chunk global store: rows 2i,2i+1 of every (b,o) ----
    #pragma unroll
    for (int k = 0; k < 16; ++k) {
        const int fid = tid + k*256;
        const int u   = fid & 7;
        const int oo  = (fid >> 3) & 63;
        const int bb  = fid >> 9;
        const float4 v = *reinterpret_cast<const float4*>(&Os[(bb*64 + oo)*36 + u*4]);
        *reinterpret_cast<float4*>(out + (size_t)(b0 + bb)*16384 + oo*256 + irow*32 + u*4) = v;
    }
}

// ---------------------------------------------------------------------------
extern "C" void kernel_launch(void* const* d_in, const int* in_sizes, int n_in,
                              void* d_out, int out_size, void* d_ws, size_t ws_size,
                              hipStream_t stream) {
    const float* x  = (const float*)d_in[0];
    const float* wu = (const float*)d_in[1];
    const float* bu = (const float*)d_in[2];
    const float* wl = (const float*)d_in[3];
    const float* bl = (const float*)d_in[4];
    const float* wr = (const float*)d_in[5];
    const float* br = (const float*)d_in[6];
    float* out = (float*)d_out;
    __bf16* Wg = (__bf16*)d_ws;   // 147456 bytes used

    kagome_prep_weights<<<288, 256, 0, stream>>>(wu, wl, wr, Wg);

    dim3 grid(256, 8, 1);         // 256 batch tiles x 8 output rows
    kagome_main<<<grid, 256, 0, stream>>>(x, bu, bl, br, Wg, out);
}
